// LocalMambaLayer_22333829940095
// MI455X (gfx1250) — matrix operand order repaired and checked by reading, not compiled, not run on hardware
//
#include <hip/hip_runtime.h>
#include <hip/hip_bf16.h>
#include <cstdint>

// ---------------------------------------------------------------------------
// Mamba2 layer for MI455X (gfx1250, wave32, WMMA, async global->LDS DMA).
//  K1: zxbcdt = x @ W_in            (bf16 WMMA, f32 accum, async LDS staging)
//  K2: dt = softplus(raw+bias), dA = exp(dt * -exp(A_log))
//  K3: depthwise conv(4) + SiLU over xBC channels
//  K4: selective scan, state in registers, 1 block per (batch, head)
//  K5: y = y*silu(z); RMSNorm (in place)
//  K6: out = x + (y @ W_out) * layer_scale   (bf16 WMMA, fused epilogue)
// ---------------------------------------------------------------------------

typedef __attribute__((ext_vector_type(16))) __bf16 v16bf;
typedef __attribute__((ext_vector_type(8)))  float  v8f;

#define MB_DMODEL   1024
#define MB_DSTATE   16
#define MB_DCONV    4
#define MB_DINNER   2048
#define MB_NHEADS   32
#define MB_HEADDIM  64
#define MB_CONVDIM  2080          // D_INNER + 2*D_STATE
#define MB_DINPROJ  4160          // 2*D_INNER + 2*D_STATE + NHEADS
#define MB_BATCH    4
#define MB_SEQLEN   2048
#define MB_ROWS     (MB_BATCH * MB_SEQLEN)   // 8192

__device__ __forceinline__ float silu_f(float v) {
    return v * (1.0f / (1.0f + __expf(-v)));
}

// Async DMA: 16 bytes global -> LDS per lane, tracked by ASYNCcnt.
// lds_off = LDS byte offset (low 32 bits of a generic shared pointer).
__device__ __forceinline__ void async_ld_b128(uint32_t lds_off, const void* g) {
    asm volatile("global_load_async_to_lds_b128 %0, %1, off"
                 :: "v"(lds_off), "v"(g) : "memory");
}
__device__ __forceinline__ void wait_async0() {
    asm volatile("s_wait_asynccnt 0x0" ::: "memory");
}

// ---------------------------------------------------------------------------
// WMMA GEMM: C[M,N] = A[M,K] * B[K,N]  (+ optional resid + per-col scale)
// block = 256 threads = 8 waves; block tile 128x64; wave tile 32x32;
// K-step 32 using v_wmma_f32_16x16x32_bf16; double-buffered async staging.
// ---------------------------------------------------------------------------
__launch_bounds__(256)
__global__ void wmma_gemm_kernel(const float* __restrict__ A,
                                 const float* __restrict__ B,
                                 float* __restrict__ C,
                                 int lda, int ldb, int ldc, int K,
                                 const float* __restrict__ resid,
                                 const float* __restrict__ scale)
{
    constexpr int BM = 128, BN = 64, BK = 32;
    constexpr int AKP = BK + 4;           // 36 floats: 16B-aligned, conflict-free
    constexpr int BNP = BN + 4;           // 68 floats: 16B-aligned
    __shared__ float sA[2][BM][AKP];      // [m][k]  (36.0 KB)
    __shared__ float sB[2][BK][BNP];      // [k][n]  (17.0 KB)

    const int tid  = threadIdx.x;
    const int wave = tid >> 5;
    const int lane = tid & 31;
    const int hl   = lane & 15;           // half-lane (row/col within 16)
    const int hi   = lane >> 4;           // which 16-lane half

    const int wm = (wave >> 1) * 32;      // wave tile origin in block (4 x 2 waves)
    const int wn = (wave & 1)  * 32;

    const int blockM = blockIdx.x * BM;
    const int blockN = blockIdx.y * BN;

    v8f acc00 = {}, acc01 = {}, acc10 = {}, acc11 = {};

    // issue the async copies for one 128x32 A tile + 32x64 B tile (6 per thread)
    auto stage = [&](int bufi, int kb) {
#pragma unroll
        for (int i = 0; i < 4; ++i) {     // A: 1024 float4 slots / 256 threads
            const int s = tid + i * 256;
            const int row = s >> 3;       // 8 float4 per k-row of 32
            const int kq  = s & 7;
            async_ld_b128((uint32_t)(uintptr_t)&sA[bufi][row][kq * 4],
                          &A[(size_t)(blockM + row) * lda + kb + kq * 4]);
        }
#pragma unroll
        for (int i = 0; i < 2; ++i) {     // B: 512 float4 slots / 256 threads
            const int s = tid + i * 256;
            const int kr = s >> 4;        // 16 float4 per n-row of 64
            const int nq = s & 15;
            async_ld_b128((uint32_t)(uintptr_t)&sB[bufi][kr][nq * 4],
                          &B[(size_t)(kb + kr) * ldb + blockN + nq * 4]);
        }
    };

    stage(0, 0);
    int buf = 0;

    for (int kb = 0; kb < K; kb += BK) {
        wait_async0();                    // this wave's tile-kb DMA landed in LDS
        __syncthreads();                  // publish to all waves; old buf now free
        if (kb + BK < K) stage(buf ^ 1, kb + BK);   // overlap DMA with WMMA below

        // ---- build bf16 fragments per ISA per-lane layouts --------------
        // A 16x32: lanes 0-15 hold K {0..7,16..23}, lanes 16-31 K {8..15,24..31}
        v16bf a0, a1, b0, b1;
        {
            const float* r0 = &sA[buf][wm + hl][0];
            const float* r1 = &sA[buf][wm + 16 + hl][0];
            const float4 p0 = *(const float4*)(r0 + hi * 8);
            const float4 p1 = *(const float4*)(r0 + hi * 8 + 4);
            const float4 p2 = *(const float4*)(r0 + 16 + hi * 8);
            const float4 p3 = *(const float4*)(r0 + 16 + hi * 8 + 4);
            a0[0]=(__bf16)p0.x; a0[1]=(__bf16)p0.y; a0[2]=(__bf16)p0.z; a0[3]=(__bf16)p0.w;
            a0[4]=(__bf16)p1.x; a0[5]=(__bf16)p1.y; a0[6]=(__bf16)p1.z; a0[7]=(__bf16)p1.w;
            a0[8]=(__bf16)p2.x; a0[9]=(__bf16)p2.y; a0[10]=(__bf16)p2.z; a0[11]=(__bf16)p2.w;
            a0[12]=(__bf16)p3.x; a0[13]=(__bf16)p3.y; a0[14]=(__bf16)p3.z; a0[15]=(__bf16)p3.w;
            const float4 q0 = *(const float4*)(r1 + hi * 8);
            const float4 q1 = *(const float4*)(r1 + hi * 8 + 4);
            const float4 q2 = *(const float4*)(r1 + 16 + hi * 8);
            const float4 q3 = *(const float4*)(r1 + 16 + hi * 8 + 4);
            a1[0]=(__bf16)q0.x; a1[1]=(__bf16)q0.y; a1[2]=(__bf16)q0.z; a1[3]=(__bf16)q0.w;
            a1[4]=(__bf16)q1.x; a1[5]=(__bf16)q1.y; a1[6]=(__bf16)q1.z; a1[7]=(__bf16)q1.w;
            a1[8]=(__bf16)q2.x; a1[9]=(__bf16)q2.y; a1[10]=(__bf16)q2.z; a1[11]=(__bf16)q2.w;
            a1[12]=(__bf16)q3.x; a1[13]=(__bf16)q3.y; a1[14]=(__bf16)q3.z; a1[15]=(__bf16)q3.w;
        }
        // B 32x16: lanes 0-15 hold K {0..7,16..23}, lanes 16-31 K {8..15,24..31}; N = hl
#pragma unroll
        for (int e = 0; e < 16; ++e) {
            b0[e] = (__bf16)sB[buf][hi * 8 + e + (e >= 8 ? 8 : 0)][wn + hl];
            b1[e] = (__bf16)sB[buf][hi * 8 + e + (e >= 8 ? 8 : 0)][wn + 16 + hl];
        }

        acc00 = __builtin_amdgcn_wmma_f32_16x16x32_bf16(false, a0, false, b0,
                                                        (short)0, acc00, false, false);
        acc01 = __builtin_amdgcn_wmma_f32_16x16x32_bf16(false, a0, false, b1,
                                                        (short)0, acc01, false, false);
        acc10 = __builtin_amdgcn_wmma_f32_16x16x32_bf16(false, a1, false, b0,
                                                        (short)0, acc10, false, false);
        acc11 = __builtin_amdgcn_wmma_f32_16x16x32_bf16(false, a1, false, b1,
                                                        (short)0, acc11, false, false);
        buf ^= 1;
    }

    // ---- epilogue: C/D layout: VGPR v -> row M = v + hi*8, col N = hl ----
    auto storeC = [&](int r, int c, float vv) {
        const size_t idx = (size_t)r * ldc + c;
        if (resid != nullptr) C[idx] = resid[idx] + vv * scale[c];
        else                  C[idx] = vv;
    };
#pragma unroll
    for (int v = 0; v < 8; ++v) {
        const int r0 = blockM + wm + v + hi * 8;
        const int r1 = r0 + 16;
        const int c0 = blockN + wn + hl;
        const int c1 = c0 + 16;
        storeC(r0, c0, acc00[v]);
        storeC(r0, c1, acc01[v]);
        storeC(r1, c0, acc10[v]);
        storeC(r1, c1, acc11[v]);
    }
}

// ---------------------------------------------------------------------------
// K2: dt = softplus(zxbcdt[...,4128+h] + dt_bias[h]); dA = exp(dt * -exp(A_log))
// ---------------------------------------------------------------------------
__global__ void dt_kernel(const float* __restrict__ zx,
                          const float* __restrict__ dt_bias,
                          const float* __restrict__ A_log,
                          float* __restrict__ dtb,
                          float* __restrict__ dAb)
{
    const int idx = blockIdx.x * blockDim.x + threadIdx.x;   // B*L*NHEADS
    const int h   = idx & (MB_NHEADS - 1);
    const int bl  = idx >> 5;
    const float raw = zx[(size_t)bl * MB_DINPROJ + (MB_DINNER + MB_CONVDIM) + h]
                      + dt_bias[h];
    const float dt  = (raw > 20.0f) ? raw : log1pf(__expf(raw));
    const float Ah  = -__expf(A_log[h]);
    dtb[idx] = dt;
    dAb[idx] = __expf(dt * Ah);
}

// ---------------------------------------------------------------------------
// K3: depthwise causal conv width-4 over xBC channels, then SiLU.
// ---------------------------------------------------------------------------
__global__ void conv_silu_kernel(const float* __restrict__ zx,
                                 const float* __restrict__ conv_w,
                                 const float* __restrict__ conv_b,
                                 float* __restrict__ xconv)
{
    const int idx = blockIdx.x * blockDim.x + threadIdx.x;   // B*L*CONV_DIM
    const int c   = idx % MB_CONVDIM;
    const int bl  = idx / MB_CONVDIM;
    const int l   = bl & (MB_SEQLEN - 1);
    float acc = conv_b[c];
#pragma unroll
    for (int k = 0; k < MB_DCONV; ++k) {
        const int t = l - (MB_DCONV - 1) + k;
        if (t >= 0) {
            acc += zx[(size_t)(bl - l + t) * MB_DINPROJ + MB_DINNER + c]
                   * conv_w[c * MB_DCONV + k];
        }
    }
    xconv[idx] = silu_f(acc);
}

// ---------------------------------------------------------------------------
// K4: selective scan. 1 block = 1 (batch, head). 256 threads; thread =
// (p = tid>>2, ng = tid&3) owns state h[p][4*ng .. 4*ng+3] in registers.
// y reduction over n=16 via two xor-shuffles inside the wave32.
// ---------------------------------------------------------------------------
__launch_bounds__(256)
__global__ void scan_kernel(const float* __restrict__ xconv,  // [B][L][2080]
                            const float* __restrict__ dtb,    // [B][L][32]
                            const float* __restrict__ dAb,    // [B][L][32]
                            const float* __restrict__ D_param,
                            float* __restrict__ ybuf)         // [B][L][2048]
{
    const int h = blockIdx.x;          // head
    const int b = blockIdx.y;          // batch
    const int tid = threadIdx.x;
    const int p  = tid >> 2;           // 0..63  (headdim)
    const int ng = tid & 3;            // n-group: owns n = 4*ng .. 4*ng+3

    float hs0 = 0.f, hs1 = 0.f, hs2 = 0.f, hs3 = 0.f;
    const float Dp = D_param[h];

    for (int t = 0; t < MB_SEQLEN; ++t) {
        const size_t rbase = ((size_t)b * MB_SEQLEN + t);
        const size_t cbase = rbase * MB_CONVDIM;
        const float xv = xconv[cbase + h * MB_HEADDIM + p];
        const float B0 = xconv[cbase + MB_DINNER + 4 * ng + 0];
        const float B1 = xconv[cbase + MB_DINNER + 4 * ng + 1];
        const float B2 = xconv[cbase + MB_DINNER + 4 * ng + 2];
        const float B3 = xconv[cbase + MB_DINNER + 4 * ng + 3];
        const float C0 = xconv[cbase + MB_DINNER + MB_DSTATE + 4 * ng + 0];
        const float C1 = xconv[cbase + MB_DINNER + MB_DSTATE + 4 * ng + 1];
        const float C2 = xconv[cbase + MB_DINNER + MB_DSTATE + 4 * ng + 2];
        const float C3 = xconv[cbase + MB_DINNER + MB_DSTATE + 4 * ng + 3];
        const float dt = dtb[rbase * MB_NHEADS + h];
        const float dA = dAb[rbase * MB_NHEADS + h];
        const float dtx = dt * xv;

        hs0 = hs0 * dA + dtx * B0;
        hs1 = hs1 * dA + dtx * B1;
        hs2 = hs2 * dA + dtx * B2;
        hs3 = hs3 * dA + dtx * B3;

        float yp = hs0 * C0 + hs1 * C1 + hs2 * C2 + hs3 * C3;
        yp += __shfl_xor(yp, 1, 32);   // reduce across the 4 n-groups
        yp += __shfl_xor(yp, 2, 32);   // (lane bits 0..1 == ng)
        if (ng == 0) {
            ybuf[rbase * MB_DINNER + h * MB_HEADDIM + p] = yp + Dp * xv;
        }
    }
}

// ---------------------------------------------------------------------------
// K5: in-place  y = y * silu(z);  y = y * rsqrt(mean(y^2)+eps) * norm_w
// 1 block per (b,l) row of 2048; 256 threads x 8 elements.
// ---------------------------------------------------------------------------
__launch_bounds__(256)
__global__ void gate_norm_kernel(float* __restrict__ ybuf,
                                 const float* __restrict__ zx,
                                 const float* __restrict__ norm_w)
{
    const int row = blockIdx.x;                       // 0..8191
    const int tid = threadIdx.x;
    const size_t ybase = (size_t)row * MB_DINNER;
    const size_t zbase = (size_t)row * MB_DINPROJ;    // z = zx[..., :2048]

    float vals[8];
    float ss = 0.f;
#pragma unroll
    for (int i = 0; i < 8; ++i) {
        const int c = tid + i * 256;
        const float v = ybuf[ybase + c] * silu_f(zx[zbase + c]);
        vals[i] = v;
        ss += v * v;
    }
    __shared__ float red[256];
    red[tid] = ss;
    __syncthreads();
    for (int s = 128; s > 0; s >>= 1) {
        if (tid < s) red[tid] += red[tid + s];
        __syncthreads();
    }
    const float inv = rsqrtf(red[0] * (1.0f / MB_DINNER) + 1e-5f);
#pragma unroll
    for (int i = 0; i < 8; ++i) {
        const int c = tid + i * 256;
        ybuf[ybase + c] = vals[i] * inv * norm_w[c];
    }
}

// ---------------------------------------------------------------------------
extern "C" void kernel_launch(void* const* d_in, const int* in_sizes, int n_in,
                              void* d_out, int out_size, void* d_ws, size_t ws_size,
                              hipStream_t stream)
{
    const float* x        = (const float*)d_in[0];   // [4,2048,1024]
    const float* W_in     = (const float*)d_in[1];   // [1024,4160]
    const float* conv_w   = (const float*)d_in[2];   // [2080,4]
    const float* conv_b   = (const float*)d_in[3];   // [2080]
    const float* dt_bias  = (const float*)d_in[4];   // [32]
    const float* A_log    = (const float*)d_in[5];   // [32]
    const float* D_param  = (const float*)d_in[6];   // [32]
    const float* norm_w   = (const float*)d_in[7];   // [2048]
    const float* W_out    = (const float*)d_in[8];   // [2048,1024]
    const float* lscale   = (const float*)d_in[9];   // [1024]
    float*       out      = (float*)d_out;           // [4,2048,1024]

    // workspace layout (bytes; 16B-aligned; total ~261 MB)
    char* ws = (char*)d_ws;
    float* zx    = (float*)(ws);                              // 8192*4160*4 = 136314880
    float* xconv = (float*)(ws + 136314880);                  // 8192*2080*4 =  68157440
    float* dtb   = (float*)(ws + 136314880 + 68157440);       // 8192*32*4   =   1048576
    float* dAb   = (float*)(ws + 136314880 + 68157440 + 1048576);
    float* ybuf  = (float*)(ws + 136314880 + 68157440 + 2 * 1048576);  // 8192*2048*4

    // K1: zxbcdt = x @ W_in      (M=8192, N=4160, K=1024)
    wmma_gemm_kernel<<<dim3(MB_ROWS / 128, MB_DINPROJ / 64), 256, 0, stream>>>(
        x, W_in, zx, MB_DMODEL, MB_DINPROJ, MB_DINPROJ, MB_DMODEL, nullptr, nullptr);

    // K2: dt / dA               (8192*32 threads)
    dt_kernel<<<(MB_ROWS * MB_NHEADS) / 256, 256, 0, stream>>>(
        zx, dt_bias, A_log, dtb, dAb);

    // K3: conv + silu           (8192*2080 threads)
    conv_silu_kernel<<<(MB_ROWS * MB_CONVDIM) / 256, 256, 0, stream>>>(
        zx, conv_w, conv_b, xconv);

    // K4: selective scan        (32 heads x 4 batches)
    scan_kernel<<<dim3(MB_NHEADS, MB_BATCH), 256, 0, stream>>>(
        xconv, dtb, dAb, D_param, ybuf);

    // K5: gate + RMSNorm (in place)
    gate_norm_kernel<<<MB_ROWS, 256, 0, stream>>>(ybuf, zx, norm_w);

    // K6: out = x + (y @ W_out) * layer_scale   (M=8192, N=1024, K=2048)
    wmma_gemm_kernel<<<dim3(MB_ROWS / 128, MB_DMODEL / 64), 256, 0, stream>>>(
        ybuf, W_out, out, MB_DINNER, MB_DMODEL, MB_DMODEL, MB_DINNER, x, lscale);
}
